// Int4PackedLinear_89361089561448
// MI455X (gfx1250) — 8-bit matrix operand order repaired and checked by reading, compiled, not run on hardware
//
#include <hip/hip_runtime.h>
#include <hip/hip_bf16.h>
#include <stdint.h>

typedef int   v8i __attribute__((ext_vector_type(8)));
typedef float v8f __attribute__((ext_vector_type(8)));
typedef int   v4i __attribute__((ext_vector_type(4)));

#define M_TOK 8192
#define K_IN  4096
#define N_OUT 4096
#define GSZ   128
#define NG    (K_IN / GSZ)

#define BM 128
#define BN 128
#define BK 128
#define LDST 144

#if defined(__AMDGCN__) && __has_builtin(__builtin_amdgcn_global_load_async_to_lds_b128)
#define ASYNC_LDS 1
typedef __attribute__((address_space(1))) v4i g_v4i;
typedef __attribute__((address_space(3))) v4i lds_v4i;
#else
#define ASYNC_LDS 0
#endif

__device__ __forceinline__ void wait_async0() {
#if __has_builtin(__builtin_amdgcn_s_wait_asynccnt)
    __builtin_amdgcn_s_wait_asynccnt(0);
#else
    asm volatile("s_wait_asynccnt 0x0" ::: "memory");
#endif
}

#if ASYNC_LDS
__device__ __forceinline__ void fill_async(const char* ga, const char* gb,
                                           char* la, char* lb) {
    g_v4i*   g1 = (g_v4i*)ga;
    g_v4i*   g2 = (g_v4i*)gb;
    lds_v4i* l1 = (lds_v4i*)la;
    lds_v4i* l2 = (lds_v4i*)lb;
    __builtin_amdgcn_global_load_async_to_lds_b128(g1, l1, 0,  0);
    __builtin_amdgcn_global_load_async_to_lds_b128(g1, l1, 16, 0);
    __builtin_amdgcn_global_load_async_to_lds_b128(g1, l1, 32, 0);
    __builtin_amdgcn_global_load_async_to_lds_b128(g1, l1, 48, 0);
    __builtin_amdgcn_global_load_async_to_lds_b128(g2, l2, 0,  0);
    __builtin_amdgcn_global_load_async_to_lds_b128(g2, l2, 16, 0);
    __builtin_amdgcn_global_load_async_to_lds_b128(g2, l2, 32, 0);
    __builtin_amdgcn_global_load_async_to_lds_b128(g2, l2, 48, 0);
}
#endif

__global__ void k_init(unsigned* amax) {
    if (threadIdx.x == 0) *amax = 0u;
}

__global__ __launch_bounds__(256)
void k_absmax(const float* __restrict__ x, unsigned* __restrict__ amax, int n4) {
    __shared__ float red[256];
    float m = 0.0f;
    const int stride = gridDim.x * blockDim.x;
    for (int i = blockIdx.x * blockDim.x + threadIdx.x; i < n4; i += stride) {
        float4 v = ((const float4*)x)[i];
        m = fmaxf(m, fmaxf(fmaxf(fabsf(v.x), fabsf(v.y)),
                           fmaxf(fabsf(v.z), fabsf(v.w))));
    }
    red[threadIdx.x] = m;
    __syncthreads();
    for (int s = 128; s > 0; s >>= 1) {
        if (threadIdx.x < s)
            red[threadIdx.x] = fmaxf(red[threadIdx.x], red[threadIdx.x + s]);
        __syncthreads();
    }
    if (threadIdx.x == 0) atomicMax(amax, __float_as_uint(red[0]));
}

__device__ __forceinline__ int q8(float v, float inv) {
    int a = (int)rintf(v * inv);
    a = a > 127 ? 127 : a;
    a = a < -127 ? -127 : a;
    return a & 255;
}

__global__ __launch_bounds__(256)
void k_quant_x(const float* __restrict__ x, const unsigned* __restrict__ amax,
               int* __restrict__ xq4, int n4) {
    const float inv = 127.0f / fmaxf(__uint_as_float(*amax), 1e-10f);
    int i = blockIdx.x * blockDim.x + threadIdx.x;
    if (i >= n4) return;
    float4 v = ((const float4*)x)[i];
    xq4[i] = q8(v.x, inv) | (q8(v.y, inv) << 8) | (q8(v.z, inv) << 16) | (q8(v.w, inv) << 24);
}

__device__ __forceinline__ int c8(int a) {
    a = a > 127 ? 127 : a;
    a = a < -127 ? -127 : a;
    return a & 255;
}

__global__ __launch_bounds__(256)
void k_pack_w(const int* __restrict__ regb, int* __restrict__ wq4, int n4) {
    int i = blockIdx.x * blockDim.x + threadIdx.x;
    if (i >= n4) return;
    int4 w = ((const int4*)regb)[i];
    wq4[i] = c8(w.x) | (c8(w.y) << 8) | (c8(w.z) << 16) | (c8(w.w) << 24);
}

__global__ __launch_bounds__(256)
void k_gemm(const char* __restrict__ xq, const char* __restrict__ wq,
            const float* __restrict__ scales, const float* __restrict__ bias,
            const unsigned* __restrict__ amax_bits, float* __restrict__ out) {
    __shared__ __align__(16) char sA[2][BM * LDST];
    __shared__ __align__(16) char sB[2][BN * LDST];

    const int tid  = threadIdx.x;
    const int lane = tid & 31;
    const int wid  = tid >> 5;
    const int wm   = (wid >> 2) * 64;
    const int wn   = (wid & 3) * 32;
    const int l15  = lane & 15;
    const int lhi  = lane >> 4;

    const int mBase = blockIdx.y * BM;
    const int nBase = blockIdx.x * BN;

    const int ldRow  = tid >> 1;
    const int ldHalf = (tid & 1) * 64;
    const char* gA = xq + (size_t)(mBase + ldRow) * K_IN + ldHalf;
    const char* gB = wq + (size_t)(nBase + ldRow) * K_IN + ldHalf;
    const int ldsOff = ldRow * LDST + ldHalf;

    v8i ai[4][2];
    v8f af[4][2];
#pragma unroll
    for (int i = 0; i < 4; ++i)
#pragma unroll
        for (int j = 0; j < 2; ++j) { ai[i][j] = 0; af[i][j] = 0.0f; }

#if ASYNC_LDS
    fill_async(gA, gB, &sA[0][0] + ldsOff, &sB[0][0] + ldsOff);
    wait_async0();
#else
#pragma unroll
    for (int q = 0; q < 4; ++q) {
        *(int4*)(&sA[0][0] + ldsOff + q * 16) = *(const int4*)(gA + q * 16);
        *(int4*)(&sB[0][0] + ldsOff + q * 16) = *(const int4*)(gB + q * 16);
    }
#endif
    __syncthreads();

    for (int kt = 0; kt < K_IN / BK; ++kt) {
        const int cur = kt & 1, nxt = cur ^ 1;
        const bool more = (kt + 1) < (K_IN / BK);

#if ASYNC_LDS
        if (more)
            fill_async(gA + (size_t)(kt + 1) * BK, gB + (size_t)(kt + 1) * BK,
                       &sA[nxt][0] + ldsOff, &sB[nxt][0] + ldsOff);
#else
        int4 pa[4], pb[4];
        if (more) {
            const char* ga = gA + (size_t)(kt + 1) * BK;
            const char* gb = gB + (size_t)(kt + 1) * BK;
#pragma unroll
            for (int q = 0; q < 4; ++q) {
                pa[q] = *(const int4*)(ga + q * 16);
                pb[q] = *(const int4*)(gb + q * 16);
            }
        }
#endif

#pragma unroll
        for (int h = 0; h < 2; ++h) {
            v8i afr[4];
#pragma unroll
            for (int i = 0; i < 4; ++i) {
                const char* p = &sA[cur][0] + (wm + i * 16 + l15) * LDST + h * 64 + lhi * 16;
#pragma unroll
                for (int q = 0; q < 4; ++q) {
                    int2 t = *(const int2*)(p + (q >> 1) * 32 + (q & 1) * 8);
                    afr[i][2 * q]     = t.x;
                    afr[i][2 * q + 1] = t.y;
                }
            }
            v8i bfr[2];
#pragma unroll
            for (int j = 0; j < 2; ++j) {
                const char* p = &sB[cur][0] + (wn + j * 16 + l15) * LDST + h * 64 + lhi * 16;
                int4 t0 = *(const int4*)(p);
                int4 t1 = *(const int4*)(p + 32);
                bfr[j][0] = t0.x; bfr[j][1] = t0.y; bfr[j][2] = t0.z; bfr[j][3] = t0.w;
                bfr[j][4] = t1.x; bfr[j][5] = t1.y; bfr[j][6] = t1.z; bfr[j][7] = t1.w;
            }
#pragma unroll
            for (int i = 0; i < 4; ++i)
#pragma unroll
                for (int j = 0; j < 2; ++j)
                    ai[i][j] = __builtin_amdgcn_wmma_i32_16x16x64_iu8(
                        true, afr[i], true, bfr[j], ai[i][j], false, false);
        }

#pragma unroll
        for (int j = 0; j < 2; ++j) {
            const float s = scales[(size_t)(nBase + wn + j * 16 + l15) * NG + kt];
#pragma unroll
            for (int i = 0; i < 4; ++i) {
#pragma unroll
                for (int e = 0; e < 8; ++e)
                    af[i][j][e] += s * (float)ai[i][j][e];
                ai[i][j] = 0;
            }
        }

#if ASYNC_LDS
        wait_async0();
#else
        if (more) {
#pragma unroll
            for (int q = 0; q < 4; ++q) {
                *(int4*)(&sA[nxt][0] + ldsOff + q * 16) = pa[q];
                *(int4*)(&sB[nxt][0] + ldsOff + q * 16) = pb[q];
            }
        }
#endif
        __syncthreads();
    }

    const float xs = fmaxf(__uint_as_float(*amax_bits), 1e-10f) / 127.0f;
#pragma unroll
    for (int j = 0; j < 2; ++j) {
        const int n = nBase + wn + j * 16 + l15;
        const float bv = bias[n];
#pragma unroll
        for (int i = 0; i < 4; ++i) {
            const int mb = mBase + wm + i * 16 + lhi * 8;
#pragma unroll
            for (int e = 0; e < 8; ++e) {
                float v = __bfloat162float(__float2bfloat16(af[i][j][e] * xs)) + bv;
                out[(size_t)(mb + e) * N_OUT + n] = v;
            }
        }
    }
}

extern "C" void kernel_launch(void* const* d_in, const int* in_sizes, int n_in,
                              void* d_out, int out_size, void* d_ws, size_t ws_size,
                              hipStream_t stream) {
    const float* x      = (const float*)d_in[0];
    const int*   regb   = (const int*)  d_in[1];
    const float* scales = (const float*)d_in[2];
    const float* bias   = (const float*)d_in[3];
    float*       out    = (float*)d_out;

    unsigned* amax = (unsigned*)d_ws;
    char*     xq   = (char*)d_ws + 256;
    char*     wqc  = xq + (size_t)M_TOK * K_IN;
    (void)ws_size; (void)n_in; (void)in_sizes; (void)out_size;

    k_init<<<1, 64, 0, stream>>>(amax);

    const int n4x = (M_TOK * K_IN) / 4;
    k_absmax<<<1024, 256, 0, stream>>>(x, amax, n4x);

    k_quant_x<<<n4x / 256, 256, 0, stream>>>(x, amax, (int*)xq, n4x);

    const int n4w = (N_OUT * K_IN) / 4;
    k_pack_w<<<n4w / 256, 256, 0, stream>>>(regb, (int*)wqc, n4w);

    dim3 grid(N_OUT / BN, M_TOK / BM);
    k_gemm<<<grid, 256, 0, stream>>>(xq, wqc, scales, bias, amax, out);
}
